// EdgeBlock_45509473468801
// MI455X (gfx1250) — matrix operand order repaired and checked by reading, not compiled, not run on hardware
//
#include <hip/hip_runtime.h>

typedef __attribute__((ext_vector_type(16))) __bf16 v16bf;
typedef __attribute__((ext_vector_type(8)))  float  v8f;
typedef __attribute__((ext_vector_type(4)))  unsigned int v4u;

union FragAB {
  v16bf bf;
  v4u   u[2];   // u[0] = VGPRs 0..3, u[1] = VGPRs 4..7
};

struct __align__(8) bf16x4 { __bf16 x, y, z, w; };

#define N_NODES 50000
#define N_EDGES 300000
#define DFEAT   256
#define K1      768        // 3*D
#define TILE_M  64         // edges per workgroup

// ---------------------------------------------------------------------------
// Prologue: W1 (768x256 f32, row-major) -> W1t (256x768 bf16, col-of-W major)
//           W2 (256x256 f32)            -> W2t (256x256 bf16)
// Column-major bf16 weights make every WMMA B-fragment two contiguous 16B loads.
// ---------------------------------------------------------------------------
__global__ __launch_bounds__(256) void convert_weights_kernel(
    const float* __restrict__ W1, const float* __restrict__ W2,
    __bf16* __restrict__ W1t, __bf16* __restrict__ W2t) {
  int idx = blockIdx.x * 256 + threadIdx.x;   // grid covers exactly 262144
  if (idx < K1 * DFEAT) {
    int k = idx / DFEAT, n = idx - k * DFEAT;
    W1t[n * K1 + k] = (__bf16)W1[idx];
  } else {
    int j = idx - K1 * DFEAT;
    int k = j / DFEAT, n = j - k * DFEAT;
    W2t[n * DFEAT + k] = (__bf16)W2[j];
  }
}

// ---------------------------------------------------------------------------
// Main fused kernel: gather -> GEMM1(bf16 WMMA) -> ReLU -> GEMM2 -> store f32
// ---------------------------------------------------------------------------
__global__ __launch_bounds__(256) void edge_mlp_kernel(
    const float* __restrict__ x_node, const float* __restrict__ x_edge,
    const int*   __restrict__ edge_index,
    const __bf16* __restrict__ W1t, const float* __restrict__ b1,
    const __bf16* __restrict__ W2t, const float* __restrict__ b2,
    float* __restrict__ out) {
  extern __shared__ __align__(16) char smem_raw[];
  __bf16* ldsA = (__bf16*)smem_raw;          // [64][768] A panel; reused as [64][256] h1

  const int tid  = threadIdx.x;
  const int lane = tid & 31;                 // wave32
  const int wave = tid >> 5;                 // 8 waves
  const int tile_base = blockIdx.x * TILE_M;

  // ---- Stage 0: gather src|dst|edge rows, f32 -> bf16, into LDS [64,768]
  for (int c = tid; c < TILE_M * 192; c += 256) {   // 192 float4 chunks / row
    int row   = c / 192;
    int chunk = c - row * 192;
    int e  = tile_base + row;
    int ec = e < N_EDGES ? e : N_EDGES - 1;
    float4 v;
    if (chunk < 64) {
      int node = edge_index[ec];                       // src
      v = ((const float4*)x_node)[node * 64 + chunk];
    } else if (chunk < 128) {
      int node = edge_index[N_EDGES + ec];             // dst
      v = ((const float4*)x_node)[node * 64 + (chunk - 64)];
    } else {
      v = ((const float4*)x_edge)[ec * 64 + (chunk - 128)];
    }
    bf16x4 p;
    p.x = (__bf16)v.x; p.y = (__bf16)v.y; p.z = (__bf16)v.z; p.w = (__bf16)v.w;
    *(bf16x4*)&ldsA[row * K1 + chunk * 4] = p;
  }
  __syncthreads();

  // Wave tiling: 4 m-tiles x 2 n-halves; each wave owns 16 rows x 128 cols.
  const int m_tile = wave & 3;
  const int n_half = wave >> 2;
  const int arow   = m_tile * 16 + (lane & 15);
  const int koffA  = (lane >> 4) << 3;   // +8  elements for lanes 16..31 (A layout)
  const int koffB  = (lane >> 4) << 3;   // +8  elements for lanes 16..31 (B layout)

  __builtin_prefetch(W1t + (n_half * 128 + (lane & 15)) * K1, 0, 0);

  const v8f vzero = {};
  v8f acc[8];
#pragma unroll
  for (int i = 0; i < 8; ++i) acc[i] = vzero;

  // ---- GEMM1: [64 x 768] x [768 x 256], 24 k-steps of 32
  for (int kt = 0; kt < K1 / 32; ++kt) {
    const int k0 = kt * 32;
    FragAB a;
    const __bf16* ap = ldsA + arow * K1 + k0 + koffA;
    a.u[0] = *(const v4u*)ap;          // K {k0..k0+7} / {k0+8..k0+15}
    a.u[1] = *(const v4u*)(ap + 16);   // K {k0+16..}  / {k0+24..}
#pragma unroll
    for (int nt = 0; nt < 8; ++nt) {
      const int col = n_half * 128 + nt * 16 + (lane & 15);
      const __bf16* bp = W1t + col * K1 + k0 + koffB;   // two 16B chunks, 32B apart
      FragAB b;
      b.u[0] = *(const v4u*)bp;
      b.u[1] = *(const v4u*)(bp + 16);
      acc[nt] = __builtin_amdgcn_wmma_f32_16x16x32_bf16(
          false, a.bf, false, b.bf, (short)0, acc[nt], false, false);
    }
  }
  __syncthreads();   // all waves done reading the A panel

  // ---- bias + ReLU, h1 -> LDS as bf16 [64, 256] (reuse A buffer)
  __bf16* ldsH = ldsA;
#pragma unroll
  for (int nt = 0; nt < 8; ++nt) {
    const int col = n_half * 128 + nt * 16 + (lane & 15);
    const float bias = b1[col];
#pragma unroll
    for (int r = 0; r < 8; ++r) {
      float h = acc[nt][r] + bias;
      h = h > 0.0f ? h : 0.0f;
      const int row = m_tile * 16 + r + ((lane >> 4) << 3);  // C layout rows
      ldsH[row * DFEAT + col] = (__bf16)h;
    }
  }
  __syncthreads();

  // ---- GEMM2: [64 x 256] x [256 x 256], 8 k-steps of 32
#pragma unroll
  for (int i = 0; i < 8; ++i) acc[i] = vzero;
  for (int kt = 0; kt < DFEAT / 32; ++kt) {
    const int k0 = kt * 32;
    FragAB a;
    const __bf16* ap = ldsH + arow * DFEAT + k0 + koffA;
    a.u[0] = *(const v4u*)ap;
    a.u[1] = *(const v4u*)(ap + 16);
#pragma unroll
    for (int nt = 0; nt < 8; ++nt) {
      const int col = n_half * 128 + nt * 16 + (lane & 15);
      const __bf16* bp = W2t + col * DFEAT + k0 + koffB;
      FragAB b;
      b.u[0] = *(const v4u*)bp;
      b.u[1] = *(const v4u*)(bp + 16);
      acc[nt] = __builtin_amdgcn_wmma_f32_16x16x32_bf16(
          false, a.bf, false, b.bf, (short)0, acc[nt], false, false);
    }
  }

  // ---- bias2 + f32 store
#pragma unroll
  for (int nt = 0; nt < 8; ++nt) {
    const int col = n_half * 128 + nt * 16 + (lane & 15);
    const float bias = b2[col];
#pragma unroll
    for (int r = 0; r < 8; ++r) {
      const int e = tile_base + m_tile * 16 + r + ((lane >> 4) << 3);
      if (e < N_EDGES)
        out[e * DFEAT + col] = acc[nt][r] + bias;
    }
  }
}

extern "C" void kernel_launch(void* const* d_in, const int* in_sizes, int n_in,
                              void* d_out, int out_size, void* d_ws, size_t ws_size,
                              hipStream_t stream) {
  const float* x_node     = (const float*)d_in[0];
  const float* x_edge     = (const float*)d_in[1];
  const int*   edge_index = (const int*)d_in[2];
  const float* W1         = (const float*)d_in[3];
  const float* b1         = (const float*)d_in[4];
  const float* W2         = (const float*)d_in[5];
  const float* b2         = (const float*)d_in[6];
  float*       out        = (float*)d_out;

  __bf16* W1t = (__bf16*)d_ws;               // 768*256 bf16 = 384 KB
  __bf16* W2t = W1t + K1 * DFEAT;            // 256*256 bf16 = 128 KB

  convert_weights_kernel<<<(K1 * DFEAT + DFEAT * DFEAT) / 256, 256, 0, stream>>>(
      W1, W2, W1t, W2t);

  const int nblocks   = (N_EDGES + TILE_M - 1) / TILE_M;   // 4688
  const size_t ldsSz  = (size_t)TILE_M * K1 * sizeof(__bf16);  // 96 KB
  edge_mlp_kernel<<<nblocks, 256, ldsSz, stream>>>(
      x_node, x_edge, edge_index, W1t, b1, W2t, b2, out);
}
